// VGG16_Cifar100_12283606466626
// MI455X (gfx1250) — 8-bit matrix operand order repaired and checked by reading, compiled, not run on hardware
//
#include <hip/hip_runtime.h>
#include <cstdint>


typedef __attribute__((ext_vector_type(8))) int v8i;

#define EPSF 1e-5f
#define BN_SLICES 8

__global__ void k_binarize(const float* __restrict__ in, int8_t* __restrict__ out, int n) {
  int i = blockIdx.x * blockDim.x + threadIdx.x;
  if (i < n) {
    float v = in[i];
    out[i] = (int8_t)((v > 0.f) - (v < 0.f));
  }
}

__global__ void k_binarize_wperm(const float* __restrict__ w, int8_t* __restrict__ out,
                                 int cinLog, int total) {
  int i = blockIdx.x * blockDim.x + threadIdx.x;
  if (i >= total) return;
  int cinMask = (1 << cinLog) - 1;
  int c  = i & cinMask;
  int t  = i >> cinLog;
  int rs = t % 9;
  int co = t / 9;
  float v = w[((size_t)((co << cinLog) + c)) * 9 + rs];
  out[i] = (int8_t)((v > 0.f) - (v < 0.f));
}

__global__ __launch_bounds__(256) void k_conv0(const float* __restrict__ x,
                                               const int8_t* __restrict__ wb,
                                               float* __restrict__ out) {
  const int H = 32, W = 32, Cin = 3;
  int idx = blockIdx.x * blockDim.x + threadIdx.x;
  int co   = idx & 63;
  int pix  = (idx >> 6) & 1023;
  int n    = idx >> 16;
  int yrow = pix >> 5;
  int xcol = pix & 31;
  float acc = 0.f;
  for (int c = 0; c < Cin; ++c)
    for (int r = 0; r < 3; ++r) {
      int yy = yrow + r - 1;
      if (yy < 0 || yy >= H) continue;
      for (int s = 0; s < 3; ++s) {
        int xx = xcol + s - 1;
        if (xx < 0 || xx >= W) continue;
        float a = x[((n * Cin + c) * H + yy) * W + xx];
        acc += a * (float)wb[co * 27 + c * 9 + r * 3 + s];
      }
    }
  out[idx] = acc;
}

template <int H, int CIN, int COUT>
__global__ __launch_bounds__(256) void k_conv_bin_wmma(
    const int8_t* __restrict__ act, const int8_t* __restrict__ wb,
    float* __restrict__ out) {
  constexpr int W      = H;
  constexpr int HW     = H * W;
  constexpr int K      = CIN * 9;
  constexpr int CBS    = CIN / 64;
  constexpr int CBLOG  = (CBS == 1) ? 0 : (CBS == 2) ? 1 : (CBS == 4) ? 2 : 3;
  constexpr int TOT    = 9 * CBS;
  constexpr int NTILES = COUT / 16;
  constexpr int NJ     = (NTILES + 7) / 8;

  const int lane = threadIdx.x & 31;
  const int wave = threadIdx.x >> 5;
  const int mt   = blockIdx.x;

  if constexpr (NTILES < 8) {
    if (wave >= NTILES) return;
  }

  const int mrow = lane & 15;
  const int hi   = lane >> 4;

  const int m    = mt * 16 + mrow;
  const int pn   = m / HW;
  const int prem = m & (HW - 1);
  const int py   = prem / W;
  const int px   = prem & (W - 1);

  v8i vzero = {};
  v8i acc[NJ];
#pragma unroll
  for (int j = 0; j < NJ; ++j) acc[j] = vzero;

  auto loadA = [&](int t) -> v8i {
    const int rs = t >> CBLOG;
    const int cb = (t & (CBS - 1)) << 6;
    const int r  = rs / 3;
    const int s  = rs - r * 3;
    const int yy = py + r - 1;
    const int xx = px + s - 1;
    const bool valid = ((unsigned)yy < (unsigned)H) & ((unsigned)xx < (unsigned)W);
    int2 w0 = {0, 0}, w1 = {0, 0}, w2 = {0, 0}, w3 = {0, 0};
    if (valid) {
      const int8_t* p =
          act + ((long)((pn * H + yy) * W + xx)) * CIN + hi * 8 + cb;
      w0 = *(const int2*)(p);
      w1 = *(const int2*)(p + 16);
      w2 = *(const int2*)(p + 32);
      w3 = *(const int2*)(p + 48);
    }
    v8i a;
    a[0] = w0.x; a[1] = w0.y; a[2] = w1.x; a[3] = w1.y;
    a[4] = w2.x; a[5] = w2.y; a[6] = w3.x; a[7] = w3.y;
    return a;
  };

  v8i a_cur = loadA(0);
  for (int t = 0; t < TOT; ++t) {
    v8i a_nxt = (t + 1 < TOT) ? loadA(t + 1) : vzero;

    const int k0 = t << 6;
#pragma unroll
    for (int j = 0; j < NJ; ++j) {
      const int nt = wave + j * 8;
      const size_t brow = (size_t)(nt * 16 + mrow) * K + k0 + hi * 8;
      v8i b;
      const int2 b0 = *(const int2*)(wb + brow);
      const int2 b1 = *(const int2*)(wb + brow + 16);
      const int2 b2 = *(const int2*)(wb + brow + 32);
      const int2 b3 = *(const int2*)(wb + brow + 48);
      b[0] = b0.x; b[1] = b0.y; b[2] = b1.x; b[3] = b1.y;
      b[4] = b2.x; b[5] = b2.y; b[6] = b3.x; b[7] = b3.y;
      acc[j] = __builtin_amdgcn_wmma_i32_16x16x64_iu8(true, a_cur, true, b,
                                                      acc[j], false, false);
    }
    a_cur = a_nxt;
  }

#pragma unroll
  for (int j = 0; j < NJ; ++j) {
    const int co = (wave + j * 8) * 16 + mrow;
#pragma unroll
    for (int d = 0; d < 8; ++d) {
      const int mm = mt * 16 + d + hi * 8;
      out[(size_t)mm * COUT + co] = (float)acc[j][d];
    }
  }
}

__global__ __launch_bounds__(256) void k_bn_stats_part(const float* __restrict__ x,
                                                       float* __restrict__ part,
                                                       int C, int rows) {
  const int tid = threadIdx.x;
  const int cl  = tid & 63;
  const int rg  = tid >> 6;
  const int c   = blockIdx.x * 64 + cl;
  const int s   = blockIdx.y;
  const int rowsPer = rows / BN_SLICES;
  __shared__ float ssum[256], ssq[256];
  float su = 0.f, q = 0.f;
  for (int rr = s * rowsPer + rg; rr < (s + 1) * rowsPer; rr += 4) {
    float v = x[(size_t)rr * C + c];
    su += v; q += v * v;
  }
  ssum[tid] = su; ssq[tid] = q;
  __syncthreads();
  if (rg == 0) {
    su = ssum[cl] + ssum[cl + 64] + ssum[cl + 128] + ssum[cl + 192];
    q  = ssq[cl]  + ssq[cl + 64]  + ssq[cl + 128]  + ssq[cl + 192];
    part[s * C + c]                 = su;
    part[BN_SLICES * C + s * C + c] = q;
  }
}

__global__ void k_bn_stats_fin(const float* __restrict__ part,
                               float* __restrict__ stats, int C) {
  int c = blockIdx.x * blockDim.x + threadIdx.x;
  if (c >= C) return;
  float su = 0.f, q = 0.f;
  for (int s = 0; s < BN_SLICES; ++s) {
    su += part[s * C + c];
    q  += part[BN_SLICES * C + s * C + c];
  }
  stats[c]     = su;
  stats[C + c] = q;
}

__global__ void k_bn_sign(const float* __restrict__ x, const float* __restrict__ stats,
                          const float* __restrict__ g, const float* __restrict__ b,
                          int8_t* __restrict__ out, int cMask, int hwLog, int total) {
  int i = blockIdx.x * blockDim.x + threadIdx.x;
  if (i >= total) return;
  int c = i & cMask;
  float cnt  = (float)(128 << hwLog);
  float mean = stats[c] / cnt;
  float var  = stats[cMask + 1 + c] / cnt - mean * mean;
  float xh   = (x[i] - mean) * rsqrtf(var + EPSF);
  float o    = g[c] * xh + b[c];
  out[i] = (int8_t)((o > 0.f) - (o < 0.f));
}

__global__ void k_maxpool_i8(const int8_t* __restrict__ in, int8_t* __restrict__ out,
                             int cLog, int hoLog, int total) {
  int i = blockIdx.x * blockDim.x + threadIdx.x;
  if (i >= total) return;
  const int Ho = 1 << hoLog;
  const int H  = Ho << 1;
  int c  = i & ((1 << cLog) - 1);
  int t  = i >> cLog;
  int xo = t & (Ho - 1);
  int yo = (t >> hoLog) & (Ho - 1);
  int n  = t >> (2 * hoLog);
  const size_t base = (((size_t)((n * H + 2 * yo) * H + 2 * xo)) << cLog) + c;
  const int dx = 1 << cLog;
  const int dy = H << cLog;
  int8_t a = in[base], b = in[base + dx], cc = in[base + dy], d = in[base + dy + dx];
  int8_t m0 = a > b ? a : b;
  int8_t m1 = cc > d ? cc : d;
  out[i] = m0 > m1 ? m0 : m1;
}

__global__ void k_fc(const int8_t* __restrict__ a, const int8_t* __restrict__ w,
                     float* __restrict__ out) {
  int i = blockIdx.x * blockDim.x + threadIdx.x;
  if (i >= 128 * 100) return;
  int col = i % 100;
  int row = i / 100;
  int s = 0;
  const int8_t* ar = a + row * 512;
  const int8_t* wr = w + col * 512;
  for (int k = 0; k < 512; ++k) s += (int)ar[k] * (int)wr[k];
  out[i] = (float)s;
}

__global__ void k_fc_stats(const float* __restrict__ h, float* __restrict__ stats) {
  int j = blockIdx.x * blockDim.x + threadIdx.x;
  if (j >= 100) return;
  float s = 0.f, q = 0.f;
  for (int r = 0; r < 128; ++r) { float v = h[r * 100 + j]; s += v; q += v * v; }
  float m = s / 128.f;
  stats[j]       = m;
  stats[100 + j] = q / 128.f - m * m;
}

__global__ __launch_bounds__(128) void k_bn1d_logsoftmax(const float* __restrict__ h,
                                                         const float* __restrict__ stats,
                                                         float* __restrict__ out) {
  int row = blockIdx.x;
  int j   = threadIdx.x;
  __shared__ float red[128];
  bool valid = (j < 100);
  float v = -1e30f;
  if (valid) v = (h[row * 100 + j] - stats[j]) * rsqrtf(stats[100 + j] + EPSF);
  red[j] = v;
  __syncthreads();
  for (int off = 64; off > 0; off >>= 1) {
    if (j < off) red[j] = fmaxf(red[j], red[j + off]);
    __syncthreads();
  }
  float vmax = red[0];
  __syncthreads();
  red[j] = valid ? __expf(v - vmax) : 0.f;
  __syncthreads();
  for (int off = 64; off > 0; off >>= 1) {
    if (j < off) red[j] += red[j + off];
    __syncthreads();
  }
  float lse = __logf(red[0]);
  if (valid) out[row * 100 + j] = v - vmax - lse;
}

extern "C" void kernel_launch(void* const* d_in, const int* in_sizes, int n_in,
                              void* d_out, int out_size, void* d_ws, size_t ws_size,
                              hipStream_t stream) {
  (void)in_sizes; (void)n_in; (void)out_size; (void)ws_size;

  const float* x = (const float*)d_in[0];
  const float* conv_w[13];
  const float* bn_g[13];
  const float* bn_b[13];
  for (int i = 0; i < 13; ++i) {
    conv_w[i] = (const float*)d_in[1 + i];
    bn_g[i]   = (const float*)d_in[27 + i];
    bn_b[i]   = (const float*)d_in[40 + i];
  }
  float* out = (float*)d_out;

  char* ws = (char*)d_ws;
  size_t off = 0;
  auto alignup = [](size_t v) { return (v + 255) & ~(size_t)255; };
  float*  conv_f = (float*)(ws + off);  off = alignup(off + (size_t)128 * 64 * 32 * 32 * 4);
  int8_t* bufA   = (int8_t*)(ws + off); off = alignup(off + (size_t)128 * 64 * 32 * 32);
  int8_t* bufB   = (int8_t*)(ws + off); off = alignup(off + (size_t)128 * 64 * 32 * 32);
  int8_t* wb     = (int8_t*)(ws + off); off = alignup(off + (size_t)512 * 4608);
  float*  part   = (float*)(ws + off);  off = alignup(off + (size_t)512 * BN_SLICES * 2 * 4);
  float*  stats  = (float*)(ws + off);  off = alignup(off + 1024 * 4);
  float*  fch    = (float*)(ws + off);  off = alignup(off + 12800 * 4);
  float*  fcs    = (float*)(ws + off);  off = alignup(off + 200 * 4);
  int8_t* fcwb   = (int8_t*)(ws + off); off = alignup(off + (size_t)100 * 512);

  const int  CH[13]   = {64, 64, 128, 128, 256, 256, 256, 512, 512, 512, 512, 512, 512};
  const bool POOL[13] = {false, true, false, true, false, false, true,
                         false, false, true, false, false, true};

  int H = 32;
  int Cin = 3;
  int8_t* cur = bufA;
  int8_t* alt = bufB;

#define LAUNCH_CONV(HH, CI, CO)                                              \
  k_conv_bin_wmma<HH, CI, CO>                                                \
      <<<dim3(128 * HH * HH / 16), 256, 0, stream>>>(cur, wb, conv_f)

  for (int i = 0; i < 13; ++i) {
    const int Cout = CH[i];
    const int wn = Cout * Cin * 9;
    if (i == 0)
      k_binarize<<<(wn + 255) / 256, 256, 0, stream>>>(conv_w[i], wb, wn);
    else
      k_binarize_wperm<<<(wn + 255) / 256, 256, 0, stream>>>(conv_w[i], wb,
                                                             __builtin_ctz(Cin), wn);

    switch (i) {
      case 0:  k_conv0<<<(128 * 64 * 1024) / 256, 256, 0, stream>>>(x, wb, conv_f); break;
      case 1:  LAUNCH_CONV(32, 64, 64);    break;
      case 2:  LAUNCH_CONV(16, 64, 128);   break;
      case 3:  LAUNCH_CONV(16, 128, 128);  break;
      case 4:  LAUNCH_CONV(8, 128, 256);   break;
      case 5:  LAUNCH_CONV(8, 256, 256);   break;
      case 6:  LAUNCH_CONV(8, 256, 256);   break;
      case 7:  LAUNCH_CONV(4, 256, 512);   break;
      case 8:  LAUNCH_CONV(4, 512, 512);   break;
      case 9:  LAUNCH_CONV(4, 512, 512);   break;
      case 10: LAUNCH_CONV(2, 512, 512);   break;
      case 11: LAUNCH_CONV(2, 512, 512);   break;
      case 12: LAUNCH_CONV(2, 512, 512);   break;
    }

    const int HW    = H * H;
    const int hwLog = __builtin_ctz(HW);
    k_bn_stats_part<<<dim3(Cout / 64, BN_SLICES), 256, 0, stream>>>(conv_f, part,
                                                                    Cout, 128 * HW);
    k_bn_stats_fin<<<(Cout + 255) / 256, 256, 0, stream>>>(part, stats, Cout);

    const int total = 128 * Cout * HW;
    k_bn_sign<<<(total + 255) / 256, 256, 0, stream>>>(conv_f, stats, bn_g[i], bn_b[i],
                                                       alt, Cout - 1, hwLog, total);
    if (POOL[i]) {
      const int Ho = H >> 1;
      const int tp = 128 * Cout * Ho * Ho;
      k_maxpool_i8<<<(tp + 255) / 256, 256, 0, stream>>>(alt, cur, __builtin_ctz(Cout),
                                                         __builtin_ctz(Ho), tp);
      H = Ho;
    } else {
      int8_t* t = cur; cur = alt; alt = t;
    }
    Cin = Cout;
  }

  k_binarize<<<(100 * 512 + 255) / 256, 256, 0, stream>>>((const float*)d_in[53], fcwb,
                                                          100 * 512);
  k_fc<<<(128 * 100 + 255) / 256, 256, 0, stream>>>(cur, fcwb, fch);
  k_fc_stats<<<1, 128, 0, stream>>>(fch, fcs);
  k_bn1d_logsoftmax<<<128, 128, 0, stream>>>(fch, fcs, out);
}
